// Per_Region_Encoding_33775622816278
// MI455X (gfx1250) — matrix operand order repaired and checked by reading, not compiled, not run on hardware
//
#include <hip/hip_runtime.h>
#include <math.h>

// ---------------------------------------------------------------------------
// CDNA5 (gfx1250) implementation: bf16 WMMA implicit-GEMM convs + LDS staging.
// Register tile per wave: 2x2 WMMA tiles (M=32 cout x N=32 pixels), so each
// pair of A/B fragment fetches feeds 4 v_wmma_f32_16x16x32_bf16 ops.
// wave32 only.
// ---------------------------------------------------------------------------

typedef __attribute__((ext_vector_type(16))) __bf16 v16bf;
typedef __attribute__((ext_vector_type(8)))  float  v8f;

#define C_CH 256   // channels everywhere

static __device__ __forceinline__ v8f wmma_bf16(v16bf a, v16bf b, v8f c) {
  // (neg_a, A, neg_b, B, c_mod, C, reuse_a, reuse_b)
  return __builtin_amdgcn_wmma_f32_16x16x32_bf16(false, a, false, b, (short)0, c,
                                                 false, false);
}

// ---------------------------------------------------------------------------
// Weight repack: fp32 conv weights -> bf16 A-fragment order
// dst[ctile(16)][tap(ntaps)][kblk(8)][lane(32)][e(16)]
// A-frag (16x32 bf16): lane = (g<<4)|m ; element e covers K = 16*g+e
// transposed=1: use w[i][o][2-ky][2-kx]  (ConvTranspose, flipped+swapped)
// ---------------------------------------------------------------------------
__global__ void k_repack(const float* __restrict__ w, __bf16* __restrict__ dst,
                         int ntaps, int transposed, long total) {
  long idx = (long)blockIdx.x * blockDim.x + threadIdx.x;
  if (idx >= total) return;
  int e     = (int)(idx & 15);
  int lane  = (int)((idx >> 4) & 31);
  int kblk  = (int)((idx >> 9) & 7);
  long hi   = idx >> 12;
  int tap   = (int)(hi % ntaps);
  int ctile = (int)(hi / ntaps);
  int m = lane & 15, g = lane >> 4;
  int o  = ctile * 16 + m;
  int kc = 16 * g + e;
  int i  = kblk * 32 + kc;
  float val;
  if (ntaps == 1) {
    val = w[o * C_CH + i];
  } else if (!transposed) {
    val = w[((o * C_CH + i) * 3 + tap / 3) * 3 + (tap % 3)];
  } else {
    val = w[((i * C_CH + o) * 3 + (2 - tap / 3)) * 3 + (2 - tap % 3)];
  }
  dst[idx] = (__bf16)val;
}

// fp32 NCHW -> bf16 [b][p][c]
__global__ void k_to_bf16T(const float* __restrict__ src, __bf16* __restrict__ dst,
                           int HW) {
  int b = blockIdx.y;
  int total = HW * C_CH;
  long base = (long)b * total;
  for (int i = blockIdx.x * blockDim.x + threadIdx.x; i < total;
       i += gridDim.x * blockDim.x) {
    int c = i & 255, p = i >> 8;
    dst[base + i] = (__bf16)src[base + (long)c * HW + p];
  }
}

__global__ void k_zero(float* __restrict__ p, int n) {
  for (int i = blockIdx.x * blockDim.x + threadIdx.x; i < n;
       i += gridDim.x * blockDim.x) p[i] = 0.f;
}

// Instance-norm stats: partial sums per (b,c), coalesced over c.
#define STAT_CHUNK 64
__global__ void k_stats_accum(const __bf16* __restrict__ act,
                              float* __restrict__ sum, float* __restrict__ sumsq,
                              int HW) {
  int b = blockIdx.y, c = threadIdx.x;
  int p0 = blockIdx.x * STAT_CHUNK;
  int p1 = p0 + STAT_CHUNK; if (p1 > HW) p1 = HW;
  float s = 0.f, s2 = 0.f;
  for (int p = p0; p < p1; ++p) {
    float v = (float)act[((long)b * HW + p) * C_CH + c];
    s += v; s2 += v * v;
  }
  atomicAdd(&sum[b * C_CH + c], s);
  atomicAdd(&sumsq[b * C_CH + c], s2);
}

__global__ void k_stats_final(const float* __restrict__ sum,
                              const float* __restrict__ sumsq,
                              float* __restrict__ mean, float* __restrict__ rstd,
                              float invHW) {
  int i = blockIdx.x * blockDim.x + threadIdx.x;
  if (i >= 4 * C_CH) return;
  float m = sum[i] * invHW;
  float v = sumsq[i] * invHW - m * m;
  mean[i] = m;
  rstd[i] = rsqrtf(v + 1e-5f);
}

__global__ void k_normalize(const __bf16* __restrict__ in,
                            const float* __restrict__ mean,
                            const float* __restrict__ rstd,
                            __bf16* __restrict__ out, int HW) {
  int b = blockIdx.y;
  const float* mb = mean + b * C_CH;
  const float* rb = rstd + b * C_CH;
  long base = (long)b * HW * C_CH;
  int total = HW * C_CH;
  for (int i = blockIdx.x * blockDim.x + threadIdx.x; i < total;
       i += gridDim.x * blockDim.x) {
    int c = i & 255;
    float v = (float)in[base + i];
    out[base + i] = (__bf16)((v - mb[c]) * rb[c]);
  }
}

// ---------------------------------------------------------------------------
// 3x3 same-conv (pad=1), bias (+optional leaky-relu 0.1), bf16 T in/out.
// Block = 4 waves; wave owns 2 cout-tiles (M=32) x 2 pixel-tiles (N=32):
// 4 WMMAs per fragment pair, 288 WMMAs per wave.
// Input strip (3 rows x 34 x 256ch = 52KB) staged in LDS.
// ---------------------------------------------------------------------------
__global__ __launch_bounds__(128)
void k_conv3x3(const __bf16* __restrict__ inT, const __bf16* __restrict__ wpack,
               const float* __restrict__ bias, __bf16* __restrict__ outT,
               int H, int W, int do_lrelu) {
  const int tpr   = W >> 5;                 // 32-pixel tiles per row
  const int tilex = blockIdx.x % tpr;
  const int oy    = blockIdx.x / tpr;
  const int b     = blockIdx.z;
  const int wave  = threadIdx.x >> 5;
  const int lane  = threadIdx.x & 31;
  const int cpair = (blockIdx.y << 2) + wave;   // 0..7 -> ctiles 2c, 2c+1
  const int xbase = tilex << 5;

  __shared__ alignas(64) __bf16 sIn[3][34][C_CH];

  const int nchunk = 3 * 34 * (C_CH / 8);
  for (int idx = threadIdx.x; idx < nchunk; idx += 128) {
    int cpart = idx & 31;
    int xx = (idx >> 5) % 34;
    int r  = (idx >> 5) / 34;
    int iy = oy - 1 + r;
    int ix = xbase - 1 + xx;
    uint4 val = make_uint4(0u, 0u, 0u, 0u);
    if (iy >= 0 && iy < H && ix >= 0 && ix < W)
      val = *(const uint4*)&inT[(((long)b * H + iy) * W + ix) * C_CH + (cpart << 3)];
    *(uint4*)&sIn[r][xx][cpart << 3] = val;
  }
  __syncthreads();

  const int g = lane >> 4, n = lane & 15;
  const int ct0 = cpair * 2;
  v8f acc[2][2] = {};
  const __bf16* wb0 = wpack + (long)(ct0 + 0) * 9 * (8 * 32 * 16);
  const __bf16* wb1 = wpack + (long)(ct0 + 1) * 9 * (8 * 32 * 16);
  for (int tap = 0; tap < 9; ++tap) {
    const int ky = tap / 3, kx = tap % 3;
    if (tap < 8) {  // prefetch next tap's weight fragments (global_prefetch_b8)
      __builtin_prefetch(wb0 + (long)(tap + 1) * (8 * 32 * 16) + lane * 16, 0, 0);
      __builtin_prefetch(wb1 + (long)(tap + 1) * (8 * 32 * 16) + lane * 16, 0, 0);
    }
    const __bf16* sB0 = &sIn[ky][n + kx][g << 4];
    const __bf16* sB1 = &sIn[ky][n + 16 + kx][g << 4];
    for (int kblk = 0; kblk < 8; ++kblk) {
      const long woff = (long)((tap * 8 + kblk) * 32 + lane) * 16;
      v16bf a0 = *(const v16bf*)(wb0 + woff);
      v16bf a1 = *(const v16bf*)(wb1 + woff);
      v16bf b0 = *(const v16bf*)(sB0 + (kblk << 5));
      v16bf b1 = *(const v16bf*)(sB1 + (kblk << 5));
      acc[0][0] = wmma_bf16(a0, b0, acc[0][0]);
      acc[0][1] = wmma_bf16(a0, b1, acc[0][1]);
      acc[1][0] = wmma_bf16(a1, b0, acc[1][0]);
      acc[1][1] = wmma_bf16(a1, b1, acc[1][1]);
    }
  }
#pragma unroll
  for (int mt = 0; mt < 2; ++mt) {
    const int cb = (ct0 + mt) * 16 + g * 8;
#pragma unroll
    for (int j = 0; j < 2; ++j) {
      long outbase = (((long)b * H + oy) * W + (xbase + 16 * j + n)) * C_CH + cb;
      __bf16 res[8];
#pragma unroll
      for (int v = 0; v < 8; ++v) {
        float f = acc[mt][j][v] + bias[cb + v];
        if (do_lrelu) f = (f >= 0.f) ? f : 0.1f * f;
        res[v] = (__bf16)f;
      }
      *(uint4*)&outT[outbase] = *(const uint4*)res;
    }
  }
}

// ---------------------------------------------------------------------------
// ConvTranspose2d k=3 s=2 p=1 op=1 via parity decomposition:
//   y[o] += wf[k] * x[(o+k-1)/2],  wf[o][i][k] = w[i][o][2-k] (packed that way).
// Even out-coord: 1 tap (k=1); odd: 2 taps (k=0,2). N-tile = 32 same-parity
// pixels in one row. 2x2 register tile. accumulate=1 adds onto outT.
// ---------------------------------------------------------------------------
__global__ __launch_bounds__(128)
void k_convT(const __bf16* __restrict__ inT, const __bf16* __restrict__ wpack,
             const float* __restrict__ bias, __bf16* __restrict__ outT,
             int Hin, int Win, int accumulate) {
  const int Hout = Hin << 1, Wout = Win << 1;
  const int tpr = Wout >> 5;                 // 32-same-parity-pixel tiles per row
  const int tile_x = blockIdx.x % tpr;
  const int oy = blockIdx.x / tpr;
  const int b = blockIdx.z;
  const int wave = threadIdx.x >> 5, lane = threadIdx.x & 31;
  const int cpair = (blockIdx.y << 2) + wave;
  const int px = tile_x & 1, t = tile_x >> 1;
  const int xbase = (t << 6) + px;           // ox = xbase + 2*(n + 16*j)
  const int py = oy & 1;
  const int rbase = py ? ((oy - 1) >> 1) : (oy >> 1);
  const int x0 = px ? ((xbase - 1) >> 1) : (xbase >> 1);

  __shared__ alignas(64) __bf16 sIn[2][33][C_CH];
  const int nchunk = 2 * 33 * (C_CH / 8);
  for (int idx = threadIdx.x; idx < nchunk; idx += 128) {
    int cpart = idx & 31;
    int xx = (idx >> 5) % 33;
    int r  = (idx >> 5) / 33;
    int iy = rbase + r, ix = x0 + xx;
    uint4 val = make_uint4(0u, 0u, 0u, 0u);
    if (iy >= 0 && iy < Hin && ix >= 0 && ix < Win)
      val = *(const uint4*)&inT[(((long)b * Hin + iy) * Win + ix) * C_CH + (cpart << 3)];
    *(uint4*)&sIn[r][xx][cpart << 3] = val;
  }
  __syncthreads();

  const int g = lane >> 4, n = lane & 15;
  const int ct0 = cpair * 2;
  v8f acc[2][2] = {};
  const __bf16* wb0 = wpack + (long)(ct0 + 0) * 9 * (8 * 32 * 16);
  const __bf16* wb1 = wpack + (long)(ct0 + 1) * 9 * (8 * 32 * 16);
  const int nky = py ? 2 : 1;
  const int nkx = px ? 2 : 1;
  for (int ti = 0; ti < nky; ++ti) {
    const int ky   = py ? (ti ? 2 : 0) : 1;
    const int rloc = py ? (ti ? 0 : 1) : 0;   // k=0 -> iy=(oy+1)/2 = rbase+1
    for (int tj = 0; tj < nkx; ++tj) {
      const int kx   = px ? (tj ? 2 : 0) : 1;
      const int xloc = px ? tj : 0;           // k=0 -> x0, k=2 -> x0+1
      const int tap = ky * 3 + kx;
      const __bf16* sB0 = &sIn[rloc][xloc + n][g << 4];
      const __bf16* sB1 = &sIn[rloc][xloc + 16 + n][g << 4];
      for (int kblk = 0; kblk < 8; ++kblk) {
        const long woff = (long)((tap * 8 + kblk) * 32 + lane) * 16;
        v16bf a0 = *(const v16bf*)(wb0 + woff);
        v16bf a1 = *(const v16bf*)(wb1 + woff);
        v16bf b0 = *(const v16bf*)(sB0 + (kblk << 5));
        v16bf b1 = *(const v16bf*)(sB1 + (kblk << 5));
        acc[0][0] = wmma_bf16(a0, b0, acc[0][0]);
        acc[0][1] = wmma_bf16(a0, b1, acc[0][1]);
        acc[1][0] = wmma_bf16(a1, b0, acc[1][0]);
        acc[1][1] = wmma_bf16(a1, b1, acc[1][1]);
      }
    }
  }
#pragma unroll
  for (int mt = 0; mt < 2; ++mt) {
    const int cb = (ct0 + mt) * 16 + g * 8;
#pragma unroll
    for (int j = 0; j < 2; ++j) {
      long outbase =
          (((long)b * Hout + oy) * Wout + (xbase + 2 * (16 * j + n))) * C_CH + cb;
      float prev[8] = {0.f, 0.f, 0.f, 0.f, 0.f, 0.f, 0.f, 0.f};
      if (accumulate) {
        uint4 pv = *(const uint4*)&outT[outbase];
        const __bf16* pb = (const __bf16*)&pv;
#pragma unroll
        for (int v = 0; v < 8; ++v) prev[v] = (float)pb[v];
      }
      __bf16 res[8];
#pragma unroll
      for (int v = 0; v < 8; ++v)
        res[v] = (__bf16)(acc[mt][j][v] + bias[cb + v] + prev[v]);
      *(uint4*)&outT[outbase] = *(const uint4*)res;
    }
  }
}

// Final 1x1 conv + tanh: GEMM K=256, 2x2 register tile (M=32 x N=32).
// Writes fp32 NCHW into d_out's x region and a bf16 T copy for pooling.
__global__ __launch_bounds__(128)
void k_conv1x1_tanh(const __bf16* __restrict__ inT, const __bf16* __restrict__ wpack,
                    const float* __restrict__ bias, float* __restrict__ outX,
                    __bf16* __restrict__ outT) {
  const int H = 256, W = 256;
  const int tpr = W >> 5;
  const int tilex = blockIdx.x % tpr;
  const int oy = blockIdx.x / tpr;
  const int b = blockIdx.z;
  const int wave = threadIdx.x >> 5, lane = threadIdx.x & 31;
  const int cpair = (blockIdx.y << 2) + wave;
  const int xbase = tilex << 5;
  const int g = lane >> 4, n = lane & 15;
  const int ct0 = cpair * 2;
  v8f acc[2][2] = {};
  const __bf16* wb0 = wpack + (long)(ct0 + 0) * (8 * 32 * 16);
  const __bf16* wb1 = wpack + (long)(ct0 + 1) * (8 * 32 * 16);
  const __bf16* sB0 = &inT[(((long)b * H + oy) * W + (xbase + n)) * C_CH + (g << 4)];
  const __bf16* sB1 = &inT[(((long)b * H + oy) * W + (xbase + 16 + n)) * C_CH + (g << 4)];
  for (int kblk = 0; kblk < 8; ++kblk) {
    const long woff = (long)(kblk * 32 + lane) * 16;
    v16bf a0 = *(const v16bf*)(wb0 + woff);
    v16bf a1 = *(const v16bf*)(wb1 + woff);
    v16bf b0 = *(const v16bf*)(sB0 + (kblk << 5));
    v16bf b1 = *(const v16bf*)(sB1 + (kblk << 5));
    acc[0][0] = wmma_bf16(a0, b0, acc[0][0]);
    acc[0][1] = wmma_bf16(a0, b1, acc[0][1]);
    acc[1][0] = wmma_bf16(a1, b0, acc[1][0]);
    acc[1][1] = wmma_bf16(a1, b1, acc[1][1]);
  }
#pragma unroll
  for (int mt = 0; mt < 2; ++mt) {
    const int cb = (ct0 + mt) * 16 + g * 8;
#pragma unroll
    for (int j = 0; j < 2; ++j) {
      long pix = (long)oy * W + xbase + 16 * j + n;
      __bf16 res[8];
#pragma unroll
      for (int v = 0; v < 8; ++v) {
        float f = tanhf(acc[mt][j][v] + bias[cb + v]);
        outX[((long)b * C_CH + cb + v) * 65536 + pix] = f;
        res[v] = (__bf16)f;
      }
      *(uint4*)&outT[((long)b * 65536 + pix) * C_CH + cb] = *(const uint4*)res;
    }
  }
}

// Region mask population counts: one block per (s,b).
__global__ void k_counts(const float* __restrict__ seg, float* __restrict__ cnt) {
  int s = blockIdx.x, b = blockIdx.y;
  const float* m = seg + ((long)b * 8 + s) * 65536;
  float acc = 0.f;
  for (int p = threadIdx.x; p < 65536; p += 256) acc += (m[p] != 0.f) ? 1.f : 0.f;
  __shared__ float red[256];
  red[threadIdx.x] = acc;
  __syncthreads();
  for (int o = 128; o > 0; o >>= 1) {
    if (threadIdx.x < o) red[threadIdx.x] += red[threadIdx.x + o];
    __syncthreads();
  }
  if (threadIdx.x == 0) cnt[b * 8 + s] = red[0];
}

// Masked segment sums + global sum: thread = channel (coalesced c-major reads).
__global__ void k_pool(const __bf16* __restrict__ xT, const float* __restrict__ seg,
                       float* __restrict__ rsum) {
  int b = blockIdx.y, c = threadIdx.x;
  int p0 = blockIdx.x * 256;
  float acc[9];
#pragma unroll
  for (int s = 0; s < 9; ++s) acc[s] = 0.f;
  for (int p = p0; p < p0 + 256; ++p) {
    float xv = (float)xT[((long)b * 65536 + p) * C_CH + c];
    acc[8] += xv;
#pragma unroll
    for (int s = 0; s < 8; ++s) {
      float mv = seg[((long)b * 8 + s) * 65536 + p];
      if (mv != 0.f) acc[s] += xv;
    }
  }
#pragma unroll
  for (int s = 0; s < 9; ++s)
    atomicAdd(&rsum[((long)b * 9 + s) * C_CH + c], acc[s]);
}

__global__ void k_finalize(const float* __restrict__ rsum,
                           const float* __restrict__ rcnt,
                           float* __restrict__ out_codes,
                           float* __restrict__ out_exist) {
  int idx = blockIdx.x * blockDim.x + threadIdx.x;
  if (idx >= 4 * 9 * C_CH) return;
  int c = idx % C_CH;
  int s = (idx / C_CH) % 9;
  int b = idx / (9 * C_CH);
  float v;
  if (s < 8) {
    float cnt = rcnt[b * 8 + s];
    v = (cnt > 0.f) ? rsum[idx] / fmaxf(cnt, 1.f) : 0.f;
    if (c == 0) out_exist[b * 8 + s] = (cnt > 0.f) ? 1.f : 0.f;
  } else {
    v = rsum[idx] * (1.f / 65536.f);
  }
  out_codes[idx] = v;
}

// ---------------------------------------------------------------------------
extern "C" void kernel_launch(void* const* d_in, const int* in_sizes, int n_in,
                              void* d_out, int out_size, void* d_ws, size_t ws_size,
                              hipStream_t stream) {
  (void)in_sizes; (void)n_in; (void)out_size; (void)ws_size;

  const float* inp = (const float*)d_in[0];
  const float* seg = (const float*)d_in[1];
  const float *c1w[3], *c1b[3], *ctw[3], *ctb[3], *shw[3], *shb[3];
  for (int k = 0; k < 3; ++k) {
    c1w[k] = (const float*)d_in[2 + 6 * k + 0];
    c1b[k] = (const float*)d_in[2 + 6 * k + 1];
    ctw[k] = (const float*)d_in[2 + 6 * k + 2];
    ctb[k] = (const float*)d_in[2 + 6 * k + 3];
    shw[k] = (const float*)d_in[2 + 6 * k + 4];
    shb[k] = (const float*)d_in[2 + 6 * k + 5];
  }
  const float* fw = (const float*)d_in[20];
  const float* fb = (const float*)d_in[21];

  // Workspace carve (~363 MB): two 128MB ping-pong activation buffers,
  // three 32MB temporaries, packed weights, stats & pooling scratch.
  char* ws = (char*)d_ws;
  __bf16* BIG0 = (__bf16*)(ws);
  __bf16* BIG1 = (__bf16*)(ws + (size_t)134217728);
  __bf16* NRM  = (__bf16*)(ws + (size_t)268435456);
  __bf16* Y1   = (__bf16*)(ws + (size_t)301989888);
  __bf16* Y1N  = (__bf16*)(ws + (size_t)335544320);
  __bf16* WP   = (__bf16*)(ws + (size_t)369098752);
  float*  sbuf = (float*)(ws + (size_t)379846656);  // sum[1024]
  float*  sqbuf = sbuf + 1024;                      // sumsq[1024]
  float*  mbuf  = sbuf + 2048;                      // mean[1024]
  float*  rbuf  = sbuf + 3072;                      // rstd[1024]
  float*  rsum  = sbuf + 4096;                      // region sums [4][9][256]
  float*  rcnt  = rsum + 9216;                      // counts [4][8]

  const long W3 = 9L * 8 * 32 * 16 * 16;  // 589824 elems per 3x3 pack
  const long W1 = 1L * 8 * 32 * 16 * 16;  // 65536 elems for 1x1 pack
  __bf16 *wp_c1[3], *wp_ct[3], *wp_sh[3];
  __bf16* p = WP;
  for (int k = 0; k < 3; ++k) {
    wp_c1[k] = p; p += W3;
    wp_ct[k] = p; p += W3;
    wp_sh[k] = p; p += W3;
  }
  __bf16* wp_f = p;

  // 1. repack weights into WMMA A-fragment order
  {
    int blocks3 = (int)((W3 + 255) / 256);
    for (int k = 0; k < 3; ++k) {
      k_repack<<<blocks3, 256, 0, stream>>>(c1w[k], wp_c1[k], 9, 0, W3);
      k_repack<<<blocks3, 256, 0, stream>>>(ctw[k], wp_ct[k], 9, 1, W3);
      k_repack<<<blocks3, 256, 0, stream>>>(shw[k], wp_sh[k], 9, 1, W3);
    }
    k_repack<<<(int)((W1 + 255) / 256), 256, 0, stream>>>(fw, wp_f, 1, 0, W1);
  }

  // 2. input fp32 NCHW -> bf16 T layout
  k_to_bf16T<<<dim3(256, 4), 256, 0, stream>>>(inp, BIG0, 32 * 32);

  // 3. three residual upsampling blocks
  for (int k = 0; k < 3; ++k) {
    int Hin = 32 << k, Win = Hin, HW = Hin * Win;
    __bf16* inb  = (k & 1) ? BIG1 : BIG0;
    __bf16* outb = (k & 1) ? BIG0 : BIG1;

    // inorm(x) -> NRM
    k_zero<<<8, 256, 0, stream>>>(sbuf, 2048);
    dim3 sgrid((HW + STAT_CHUNK - 1) / STAT_CHUNK, 4);
    k_stats_accum<<<sgrid, 256, 0, stream>>>(inb, sbuf, sqbuf, HW);
    k_stats_final<<<4, 256, 0, stream>>>(sbuf, sqbuf, mbuf, rbuf, 1.0f / HW);
    k_normalize<<<dim3(1024, 4), 256, 0, stream>>>(inb, mbuf, rbuf, NRM, HW);

    // conv1 3x3 + bias + lrelu -> Y1
    dim3 cgrid((Win / 32) * Hin, 2, 4);
    k_conv3x3<<<cgrid, 128, 0, stream>>>(NRM, wp_c1[k], c1b[k], Y1, Hin, Win, 1);

    // inorm(h) -> Y1N
    k_zero<<<8, 256, 0, stream>>>(sbuf, 2048);
    k_stats_accum<<<sgrid, 256, 0, stream>>>(Y1, sbuf, sqbuf, HW);
    k_stats_final<<<4, 256, 0, stream>>>(sbuf, sqbuf, mbuf, rbuf, 1.0f / HW);
    k_normalize<<<dim3(1024, 4), 256, 0, stream>>>(Y1, mbuf, rbuf, Y1N, HW);

    // convT(h) -> out ; shortcut convT(x) accumulates onto out
    dim3 tgrid(((2 * Win) / 32) * (2 * Hin), 2, 4);
    k_convT<<<tgrid, 128, 0, stream>>>(Y1N, wp_ct[k], ctb[k], outb, Hin, Win, 0);
    k_convT<<<tgrid, 128, 0, stream>>>(inb, wp_sh[k], shb[k], outb, Hin, Win, 1);
  }

  float* out_codes = (float*)d_out;       // (4,9,256)
  float* out_exist = out_codes + 9216;    // (4,8)
  float* out_x     = out_codes + 9248;    // (4,256,256,256) NCHW

  // 4. final 1x1 conv + tanh (block-3 output lives in BIG1; bf16 copy -> BIG0)
  k_conv1x1_tanh<<<dim3(8 * 256, 2, 4), 128, 0, stream>>>(BIG1, wp_f, fb,
                                                          out_x, BIG0);

  // 5. per-region encoding (seg is already 256x256 -> nearest resize = identity)
  k_zero<<<40, 256, 0, stream>>>(rsum, 9216 + 32);
  k_counts<<<dim3(8, 4), 256, 0, stream>>>(seg, rcnt);
  k_pool<<<dim3(256, 4), 256, 0, stream>>>(BIG0, seg, rsum);
  k_finalize<<<36, 256, 0, stream>>>(rsum, rcnt, out_codes, out_exist);
}
